// Encoder2_79087527789132
// MI455X (gfx1250) — matrix operand order repaired and checked by reading, not compiled, not run on hardware
//
#include <hip/hip_runtime.h>
#include <hip/hip_bf16.h>

#define NN      10000      // nodes
#define NE      80000      // edges
#define EPSB    1e-5f

typedef __attribute__((ext_vector_type(16))) _Float16 v16h;
typedef __attribute__((ext_vector_type(8)))  _Float16 v8h;
typedef __attribute__((ext_vector_type(8)))  float    v8f;

// ---------------- WMMA fragment loaders (wave32, 16x16x32 f16) ----------------
// A is row-major [M,K] f16. A layout (matches B): lane<16 holds K 0..15, lane>=16 holds K 16..31, row = lane&15.
__device__ __forceinline__ v16h load_a_frag(const _Float16* __restrict__ A, int lda,
                                            int row0, int k0, int lane) {
  int m    = lane & 15;
  int koff = (lane < 16) ? 0 : 16;
  const _Float16* p = A + (size_t)(row0 + m) * lda + (k0 + koff);
  v8h lo = *(const v8h*)p;          // K = k0+koff .. +7
  v8h hi = *(const v8h*)(p + 8);    // K = k0+koff+8 .. +15
  v16h r;
#pragma unroll
  for (int j = 0; j < 8; ++j) { r[j] = lo[j]; r[8 + j] = hi[j]; }
  return r;
}
// Bt is transposed weights [N,K] f16 (each output column's K contiguous).
// B layout used here (same as A): col = lane&15; lane<16 holds K 0..15, lane>=16 holds K 16..31.
__device__ __forceinline__ v16h load_b_frag(const _Float16* __restrict__ Bt, int ldb,
                                            int col0, int k0, int lane) {
  int n    = lane & 15;
  int koff = (lane < 16) ? 0 : 16;
  return *(const v16h*)(Bt + (size_t)(col0 + n) * ldb + (k0 + koff));
}

// ---------------- generic WMMA GEMM: C[M,N] = A16[M,K] * Bt16[N,K]^T (+bias) ----------------
__global__ void k_gemm16(const _Float16* __restrict__ A, const _Float16* __restrict__ Bt,
                         const float* __restrict__ bias, float* __restrict__ C,
                         int M, int N, int K) {
  int lane = threadIdx.x & 31;
  int wave = threadIdx.x >> 5;
  int tiles_n = N >> 4;
  int tile = blockIdx.x * (blockDim.x >> 5) + wave;
  if (tile >= (M >> 4) * tiles_n) return;           // whole-wave uniform: EXEC stays all-ones
  int row0 = (tile / tiles_n) << 4;
  int col0 = (tile % tiles_n) << 4;
  v8f acc = {};
  for (int k0 = 0; k0 < K; k0 += 32) {
    v16h a = load_a_frag(A, K, row0, k0, lane);
    v16h b = load_b_frag(Bt, K, col0, k0, lane);
    acc = __builtin_amdgcn_wmma_f32_16x16x32_f16(false, a, false, b, (short)0, acc, false, false);
  }
  int n_    = lane & 15;
  int m_off = (lane < 16) ? 0 : 8;
  int col   = col0 + n_;
  float bv  = bias ? bias[col] : 0.f;
#pragma unroll
  for (int r = 0; r < 8; ++r)
    C[(size_t)(row0 + r + m_off) * N + col] = acc[r] + bv;
}

// ---------- fused NNConv tail: sigmoid(h@Wout+b) einsum n_feat[src], atomic scatter to dst ----------
// A-tile (16 edges x 256 K of H16) is loop-invariant across the 7 input features:
// hold all 8 A-fragments in registers (64 VGPRs) and stream only B through the i/k loops.
__global__ void k_fused_msg(const _Float16* __restrict__ H16,   // [E,256]
                            const _Float16* __restrict__ WoutT, // [1792,256]
                            const float* __restrict__ bout,     // [1792]
                            const float* __restrict__ n_feat,   // [NN,7]
                            const int* __restrict__ src, const int* __restrict__ dst,
                            float* __restrict__ xsum) {         // [NN,256]
  __shared__ float s_xs[8][16 * 7];
  __shared__ int   s_dst[8][16];
  int lane = threadIdx.x & 31;
  int w    = threadIdx.x >> 5;
  int tile = blockIdx.x * 8 + w;                 // (NE/16)*16 = 80000 tiles
  if (tile >= (NE >> 4) * 16) return;
  int row0 = (tile >> 4) << 4;
  int col0 = (tile & 15) << 4;
  for (int t = lane; t < 16; t += 32) s_dst[w][t] = dst[row0 + t];
  for (int t = lane; t < 112; t += 32) {
    int m = t / 7, i = t % 7;
    s_xs[w][t] = n_feat[(size_t)src[row0 + m] * 7 + i];
  }
  // load A-tile once (8 fragments covering K=0..255)
  v16h afrag[8];
#pragma unroll
  for (int k0 = 0; k0 < 256; k0 += 32)
    afrag[k0 >> 5] = load_a_frag(H16, 256, row0, k0, lane);

  int n_    = lane & 15;
  int m_off = (lane < 16) ? 0 : 8;
  v8f acc = {};
#pragma unroll 1
  for (int i = 0; i < 7; ++i) {
    v8f t = {};
    int bcol = i * 256 + col0;                   // Wout column block for input feature i
#pragma unroll
    for (int k0 = 0; k0 < 256; k0 += 32) {
      v16h b = load_b_frag(WoutT, 256, bcol, k0, lane);
      t = __builtin_amdgcn_wmma_f32_16x16x32_f16(false, afrag[k0 >> 5], false, b, (short)0, t, false, false);
    }
    float bv = bout[bcol + n_];
#pragma unroll
    for (int r = 0; r < 8; ++r) {
      float z  = t[r] + bv;
      float sg = 1.f / (1.f + __expf(-z));
      acc[r] += s_xs[w][(r + m_off) * 7 + i] * sg;
    }
  }
#pragma unroll
  for (int r = 0; r < 8; ++r)
    atomicAdd(&xsum[(size_t)s_dst[w][r + m_off] * 256 + (col0 + n_)], acc[r]);
}

// ---------------- small/elementwise kernels ----------------
__global__ void k_fill(float* p, long long n, float v) {
  long long i = (long long)blockIdx.x * blockDim.x + threadIdx.x;
  if (i < n) p[i] = v;
}
// f32 [K,N] weight -> f16 transposed [Npad,K] (zero-padded columns)
__global__ void k_convw(const float* __restrict__ W, _Float16* __restrict__ Bt,
                        int K, int N, int Npad) {
  int i = blockIdx.x * blockDim.x + threadIdx.x;
  if (i >= Npad * K) return;
  int n = i / K, k = i % K;
  Bt[i] = (n < N) ? (_Float16)W[(size_t)k * N + n] : (_Float16)0.f;
}
__global__ void k_layer0(const float* __restrict__ ef, const float* __restrict__ W,
                         const float* __restrict__ b, float* __restrict__ Z) {
  int i = blockIdx.x * blockDim.x + threadIdx.x;     // NE*32
  if (i >= NE * 32) return;
  int e = i >> 5, o = i & 31;
  float s = b[o];
#pragma unroll
  for (int k = 0; k < 3; ++k) s += ef[e * 3 + k] * W[k * 32 + o];
  Z[i] = s;
}
__global__ void k_colstats(const float* __restrict__ Z, float* __restrict__ sum,
                           float* __restrict__ sq, int M, int d) {
  int r0 = blockIdx.x * 256;
  int r1 = r0 + 256 < M ? r0 + 256 : M;
  for (int c = threadIdx.x; c < d; c += blockDim.x) {
    float s = 0.f, q = 0.f;
    for (int r = r0; r < r1; ++r) { float v = Z[(size_t)r * d + c]; s += v; q += v * v; }
    atomicAdd(&sum[c], s); atomicAdd(&sq[c], q);
  }
}
__global__ void k_bnrelu(const float* __restrict__ Z, const float* __restrict__ sum,
                         const float* __restrict__ sq, const float* __restrict__ g,
                         const float* __restrict__ be, _Float16* __restrict__ H,
                         int M, int d) {
  int i = blockIdx.x * blockDim.x + threadIdx.x;
  if (i >= M * d) return;
  int c = i % d;
  float m = sum[c] / (float)M;
  float v = sq[c] / (float)M - m * m;
  float x = (Z[i] - m) * rsqrtf(v + EPSB) * g[c] + be[c];
  H[i] = (_Float16)fmaxf(x, 0.f);
}
__global__ void k_deg(const int* __restrict__ dst, float* __restrict__ deg) {
  int i = blockIdx.x * blockDim.x + threadIdx.x;
  if (i < NE) atomicAdd(&deg[dst[i]], 1.f);
}
__global__ void k_finalize_x(const float* __restrict__ xsum, const float* __restrict__ deg,
                             const float* __restrict__ nnb, float* __restrict__ X,
                             _Float16* __restrict__ X16) {
  int i = blockIdx.x * blockDim.x + threadIdx.x;     // NN*256
  if (i >= NN * 256) return;
  int n = i >> 8, o = i & 255;
  float v = xsum[i] / fmaxf(deg[n], 1.f) + nnb[o];
  X[i] = v; X16[i] = (_Float16)v;
}
__global__ void k_diff16(const float* __restrict__ X, const int* __restrict__ src,
                         const int* __restrict__ dst, _Float16* __restrict__ D, int K) {
  int i = blockIdx.x * blockDim.x + threadIdx.x;     // NE*K
  if (i >= NE * K) return;
  int e = i / K, k = i % K;
  D[i] = (_Float16)(X[(size_t)dst[e] * K + k] - X[(size_t)src[e] * K + k]);
}
__global__ void k_combine(float* __restrict__ Z, const float* __restrict__ PHI,
                          const int* __restrict__ dst, const float* __restrict__ bt,
                          const float* __restrict__ bp, int d, int dreal) {
  int i = blockIdx.x * blockDim.x + threadIdx.x;     // NE*d
  if (i >= NE * d) return;
  int e = i / d, o = i % d;
  float add = PHI[(size_t)dst[e] * d + o];
  if (o < dreal) add += bt[o] + bp[o];
  Z[i] += add;
}
__device__ __forceinline__ void atomicMaxF(float* addr, float val) {
  if (!(val < 0.f)) atomicMax((int*)addr, __float_as_int(val));
  else              atomicMin((unsigned int*)addr, (unsigned int)__float_as_int(val));
}
__global__ void k_bn_max(const float* __restrict__ Z, const float* __restrict__ sum,
                         const float* __restrict__ sq, const float* __restrict__ g,
                         const float* __restrict__ be, const int* __restrict__ dst,
                         float* __restrict__ xn, int d, int dreal) {
  int i = blockIdx.x * blockDim.x + threadIdx.x;     // NE*d
  if (i >= NE * d) return;
  int e = i / d, o = i % d;
  float m  = sum[o] / (float)NE;
  float v  = sq[o] / (float)NE - m * m;
  float gv = (o < dreal) ? g[o]  : 1.f;
  float bv = (o < dreal) ? be[o] : 0.f;
  float val = (Z[i] - m) * rsqrtf(v + EPSB) * gv + bv;
  atomicMaxF(&xn[(size_t)dst[e] * d + o], val);
}
__global__ void k_final_max(const float* __restrict__ xn, const float* __restrict__ deg,
                            float* __restrict__ X, _Float16* __restrict__ X16,
                            float* __restrict__ out, int d, int dreal) {
  int i = blockIdx.x * blockDim.x + threadIdx.x;     // NN*d
  if (i >= NN * d) return;
  int n = i / d, o = i % d;
  float v = (deg[n] > 0.f) ? xn[i] : 0.f;
  X[i] = v; X16[i] = (_Float16)v;
  if (out && o < dreal) out[(size_t)n * dreal + o] = v;
}

// ---------------- host orchestration ----------------
extern "C" void kernel_launch(void* const* d_in, const int* in_sizes, int n_in,
                              void* d_out, int out_size, void* d_ws, size_t ws_size,
                              hipStream_t stream) {
  const float* n_feat = (const float*)d_in[0];
  const float* e_feat = (const float*)d_in[1];
  const int*   src    = (const int*)d_in[2];
  const int*   dst    = (const int*)d_in[3];
  const float *mW[4], *mB[4], *mG[4], *mBe[4];
  for (int l = 0; l < 4; ++l) {
    mW[l]  = (const float*)d_in[4 + 4 * l];
    mB[l]  = (const float*)d_in[5 + 4 * l];
    mG[l]  = (const float*)d_in[6 + 4 * l];
    mBe[l] = (const float*)d_in[7 + 4 * l];
  }
  const float* Wout = (const float*)d_in[20];
  const float* bout = (const float*)d_in[21];
  const float* nnb  = (const float*)d_in[22];
  const float *eWt[4], *eBt[4], *eWp[4], *eBp[4], *eG[4], *eBe[4];
  for (int l = 0; l < 4; ++l) {
    eWt[l] = (const float*)d_in[23 + 6 * l]; eBt[l] = (const float*)d_in[24 + 6 * l];
    eWp[l] = (const float*)d_in[25 + 6 * l]; eBp[l] = (const float*)d_in[26 + 6 * l];
    eG[l]  = (const float*)d_in[27 + 6 * l]; eBe[l] = (const float*)d_in[28 + 6 * l];
  }

  // workspace layout (256B aligned)
  char* base = (char*)d_ws; size_t off = 0;
  auto alloc = [&](size_t bytes) -> void* {
    void* p = base + off; off = (off + bytes + 255) & ~(size_t)255; return p;
  };
  float*     Z    = (float*)    alloc((size_t)NE * 256 * 4);
  _Float16*  H16  = (_Float16*) alloc((size_t)NE * 256 * 2);
  float*     PHI  = (float*)    alloc((size_t)NN * 256 * 4);
  float*     XS   = (float*)    alloc((size_t)NN * 256 * 4);   // xsum / xmax
  float*     X    = (float*)    alloc((size_t)NN * 256 * 4);
  _Float16*  X16  = (_Float16*) alloc((size_t)NN * 256 * 2);
  float*     DEG  = (float*)    alloc((size_t)NN * 4);
  float*     SUM  = (float*)    alloc(512 * 4);                // [256 sum | 256 sumsq]
  float*     SQ   = SUM + 256;
  _Float16*  W1T  = (_Float16*) alloc(64 * 32 * 2);
  _Float16*  W2T  = (_Float16*) alloc(128 * 64 * 2);
  _Float16*  W3T  = (_Float16*) alloc(256 * 128 * 2);
  _Float16*  WOT  = (_Float16*) alloc((size_t)1792 * 256 * 2);
  int Kd[4] = {256, 128, 64, 32}, Nd[4] = {128, 64, 32, 8}, Np[4] = {128, 64, 32, 16};
  _Float16 *WtT[4], *WpT[4];
  for (int l = 0; l < 4; ++l) {
    WtT[l] = (_Float16*)alloc((size_t)Np[l] * Kd[l] * 2);
    WpT[l] = (_Float16*)alloc((size_t)Np[l] * Kd[l] * 2);
  }

  auto g1 = [](long long n) { return dim3((unsigned)((n + 255) / 256)); };

  // ---- weight conversion (f32 -> transposed f16) ----
  k_convw<<<g1(64 * 32), 256, 0, stream>>>(mW[1], W1T, 32, 64, 64);
  k_convw<<<g1(128 * 64), 256, 0, stream>>>(mW[2], W2T, 64, 128, 128);
  k_convw<<<g1(256 * 128), 256, 0, stream>>>(mW[3], W3T, 128, 256, 256);
  k_convw<<<g1((long long)1792 * 256), 256, 0, stream>>>(Wout, WOT, 256, 1792, 1792);
  for (int l = 0; l < 4; ++l) {
    k_convw<<<g1(Np[l] * Kd[l]), 256, 0, stream>>>(eWt[l], WtT[l], Kd[l], Nd[l], Np[l]);
    k_convw<<<g1(Np[l] * Kd[l]), 256, 0, stream>>>(eWp[l], WpT[l], Kd[l], Nd[l], Np[l]);
  }

  // ---- edge MLP layer 0 (3->32, scalar) ----
  k_layer0<<<g1((long long)NE * 32), 256, 0, stream>>>(e_feat, mW[0], mB[0], Z);
  k_fill<<<g1(512), 256, 0, stream>>>(SUM, 512, 0.f);
  k_colstats<<<dim3((NE + 255) / 256), 256, 0, stream>>>(Z, SUM, SQ, NE, 32);
  k_bnrelu<<<g1((long long)NE * 32), 256, 0, stream>>>(Z, SUM, SQ, mG[0], mBe[0], H16, NE, 32);

  // ---- MLP layers 1..3 (WMMA GEMMs) ----
  const _Float16* WT[3] = {W1T, W2T, W3T};
  int mlpK[3] = {32, 64, 128}, mlpN[3] = {64, 128, 256};
  for (int l = 0; l < 3; ++l) {
    int tiles = (NE / 16) * (mlpN[l] / 16);
    k_gemm16<<<dim3((tiles + 7) / 8), 256, 0, stream>>>(H16, WT[l], mB[l + 1], Z, NE, mlpN[l], mlpK[l]);
    k_fill<<<g1(512), 256, 0, stream>>>(SUM, 512, 0.f);
    k_colstats<<<dim3((NE + 255) / 256), 256, 0, stream>>>(Z, SUM, SQ, NE, mlpN[l]);
    k_bnrelu<<<g1((long long)NE * mlpN[l]), 256, 0, stream>>>(Z, SUM, SQ, mG[l + 1], mBe[l + 1], H16, NE, mlpN[l]);
  }

  // ---- NNConv: fused sigmoid(h@Wout)+einsum, segment-sum, degree, finalize ----
  k_fill<<<g1((long long)NN * 256), 256, 0, stream>>>(XS, (long long)NN * 256, 0.f);
  k_fill<<<g1(NN), 256, 0, stream>>>(DEG, NN, 0.f);
  k_deg<<<g1(NE), 256, 0, stream>>>(dst, DEG);
  k_fused_msg<<<dim3((NE / 16) * 16 / 8), 256, 0, stream>>>(H16, WOT, bout, n_feat, src, dst, XS);
  k_finalize_x<<<g1((long long)NN * 256), 256, 0, stream>>>(XS, DEG, nnb, X, X16);

  // ---- EdgeConv stack ----
  for (int l = 0; l < 4; ++l) {
    int K = Kd[l], d = Np[l], dreal = Nd[l];
    k_diff16<<<g1((long long)NE * K), 256, 0, stream>>>(X, src, dst, H16, K);
    int tilesT = (NE / 16) * (d / 16);
    k_gemm16<<<dim3((tilesT + 7) / 8), 256, 0, stream>>>(H16, WtT[l], nullptr, Z, NE, d, K);
    int tilesP = (NN / 16) * (d / 16);
    k_gemm16<<<dim3((tilesP + 7) / 8), 256, 0, stream>>>(X16, WpT[l], nullptr, PHI, NN, d, K);
    k_combine<<<g1((long long)NE * d), 256, 0, stream>>>(Z, PHI, dst, eBt[l], eBp[l], d, dreal);
    k_fill<<<g1(512), 256, 0, stream>>>(SUM, 512, 0.f);
    k_colstats<<<dim3((NE + 255) / 256), 256, 0, stream>>>(Z, SUM, SQ, NE, d);
    k_fill<<<g1((long long)NN * d), 256, 0, stream>>>(XS, (long long)NN * d, -3.402823466e38f);
    k_bn_max<<<g1((long long)NE * d), 256, 0, stream>>>(Z, SUM, SQ, eG[l], eBe[l], dst, XS, d, dreal);
    k_final_max<<<g1((long long)NN * d), 256, 0, stream>>>(XS, DEG, X, X16,
                                                           (l == 3) ? (float*)d_out : nullptr, d, dreal);
  }
}
